// Block_74363063763569
// MI455X (gfx1250) — compile-verified, matrix operand order checked by reading, not run on hardware
//
#include <hip/hip_runtime.h>
#include <hip/hip_bf16.h>
#include <stdint.h>

#define DIM 768
#define HEADS 12
#define HEAD_DIM 64
#define NTOK 197
#define BATCH 64
#define HIDDEN 3072
#define NROWS (BATCH * NTOK)   // 12608 = 788 * 16
#define MTILES (NROWS / 16)    // 788 (even -> 394 blocks of 32 rows)
#define NKPAD 224              // keys padded: 13*16=208 -> 224 = 7*32
#define QSCALE 0.125f          // 64^-0.5
#define KCHUNK 64              // K columns staged per LDS chunk (2 WMMA k-steps)

#if defined(__gfx1250__) && __has_builtin(__builtin_amdgcn_global_load_async_to_lds_b128) && __has_builtin(__builtin_amdgcn_s_wait_asynccnt)
#define ASYNC_LDS 1
#else
#define ASYNC_LDS 0
#endif

typedef int v4i_b128 __attribute__((vector_size(16)));   // builtin's pointee type

typedef __attribute__((ext_vector_type(16))) __bf16 v16bf;
typedef __attribute__((ext_vector_type(8)))  __bf16 v8bf;
typedef __attribute__((ext_vector_type(8)))  float  v8f;

#if ASYNC_LDS
__device__ __forceinline__ void async_b128(const __bf16* g, __bf16* l) {
    __builtin_amdgcn_global_load_async_to_lds_b128(
        (v4i_b128*)(uintptr_t)g, (v4i_b128*)l, 0, 0);
}
#endif

// Concatenate two contiguous 8x bf16 chunks into a 16-wide WMMA fragment.
// A-fragment (16x32, lane m holds row M=m): elems 0..7 -> K = g*8+e,
// elems 8..15 -> K = 16+g*8+e (g = lane>>4). B-fragment mirrors with N per lane.
__device__ __forceinline__ v16bf frag_cat(const __bf16* lo, const __bf16* hi) {
    v8bf a = *(const v8bf*)lo;
    v8bf b = *(const v8bf*)hi;
    return __builtin_shufflevector(a, b, 0,1,2,3,4,5,6,7,8,9,10,11,12,13,14,15);
}

__device__ __forceinline__ v8f wmma_bf16(v16bf a, v16bf b, v8f c) {
    return __builtin_amdgcn_wmma_f32_16x16x32_bf16(false, a, false, b, (short)0, c,
                                                   false, false);
}

// Block = 256 threads = 8 waves covering a 32-row x 256-col output tile.
// Each wave owns 2 M-tiles x 2 N-tiles (4 WMMA accumulators). The 32xKCHUNK
// A panel is staged in LDS. Preferred path: triple-buffered
// GLOBAL_LOAD_ASYNC_TO_LDS_B128 (ASYNCcnt), one barrier per chunk; fallback:
// double-buffered register staging. A: [M,K] bf16 row-major; W: [Nout,K] bf16
// row-major (computes A @ W^T).
__device__ __forceinline__ void gemm_rowblock32(const __bf16* __restrict__ A,
                                                const __bf16* __restrict__ W,
                                                int K, int m0, int n0w,
                                                __bf16* lds, v8f acc[2][2]) {
    const int t    = threadIdx.x;
    const int lane = t & 31;
    const int m    = lane & 15;
    const int g    = lane >> 4;
    const int srow = t >> 3;          // 32 rows, 8 threads/row
    const int scol = (t & 7) * 8;     // 8 bf16 (16B) per thread
    v8f z = {0.f, 0.f, 0.f, 0.f, 0.f, 0.f, 0.f, 0.f};
    acc[0][0] = z; acc[0][1] = z; acc[1][0] = z; acc[1][1] = z;

    const __bf16* Ag  = A + ((size_t)m0 + srow) * K + scol;  // per-thread src
    const __bf16* Wr0 = W + (size_t)(n0w + m) * K;
    const __bf16* Wr1 = W + (size_t)(n0w + 16 + m) * K;
    const int nchunks = K / KCHUNK;   // >= 12 for all call sites

#if ASYNC_LDS
    const int BUFE = 32 * KCHUNK;
    __bf16* wr0 = lds + srow * KCHUNK + scol;   // this thread's slot, buffer 0
    __bf16* wr1 = wr0 + BUFE;
    __bf16* wr2 = wr0 + 2 * BUFE;
    const __bf16* rd0 = lds;
    const __bf16* rd1 = lds + BUFE;
    const __bf16* rd2 = lds + 2 * BUFE;
    async_b128(Ag, wr0);
    async_b128(Ag + KCHUNK, wr1);
    for (int c = 0; c < nchunks; ++c) {
        if (c + 1 < nchunks) __builtin_amdgcn_s_wait_asynccnt(1);  // chunk c landed
        else                 __builtin_amdgcn_s_wait_asynccnt(0);
        __syncthreads();                       // all waves' chunk c visible
        const int k0 = c * KCHUNK;
        if (k0 + 2 * KCHUNK < K) {             // warm L2/L0 for B
            __builtin_prefetch(Wr0 + k0 + 2 * KCHUNK, 0, 1);
            __builtin_prefetch(Wr1 + k0 + 2 * KCHUNK, 0, 1);
        }
        #pragma unroll
        for (int kk = 0; kk < KCHUNK; kk += 32) {
            v16bf a0 = frag_cat(&rd0[m * KCHUNK + kk + g * 8],
                                &rd0[m * KCHUNK + kk + 16 + g * 8]);
            v16bf a1 = frag_cat(&rd0[(16 + m) * KCHUNK + kk + g * 8],
                                &rd0[(16 + m) * KCHUNK + kk + 16 + g * 8]);
            v16bf b0 = frag_cat(Wr0 + k0 + kk + g * 8, Wr0 + k0 + kk + 16 + g * 8);
            v16bf b1 = frag_cat(Wr1 + k0 + kk + g * 8, Wr1 + k0 + kk + 16 + g * 8);
            acc[0][0] = wmma_bf16(a0, b0, acc[0][0]);
            acc[1][0] = wmma_bf16(a1, b0, acc[1][0]);
            acc[0][1] = wmma_bf16(a0, b1, acc[0][1]);
            acc[1][1] = wmma_bf16(a1, b1, acc[1][1]);
        }
        // Refill the buffer that held chunk c-1 (reads finished before this
        // iteration's barrier -> safe) with chunk c+2.
        if (c + 2 < nchunks)
            async_b128(Ag + (size_t)(c + 2) * KCHUNK, wr2);
        const __bf16* rt = rd0; rd0 = rd1; rd1 = rd2; rd2 = rt;
        __bf16*       wt = wr0; wr0 = wr1; wr1 = wr2; wr2 = wt;
    }
#else
    __bf16* buf0 = lds;
    __bf16* buf1 = lds + 32 * KCHUNK;
    v8bf stage = *(const v8bf*)Ag;                          // chunk 0
    *(v8bf*)&buf0[srow * KCHUNK + scol] = stage;
    if (nchunks > 1) stage = *(const v8bf*)(Ag + KCHUNK);   // chunk 1 in regs
    __syncthreads();
    for (int c = 0; c < nchunks; ++c) {
        const __bf16* cur = (c & 1) ? buf1 : buf0;
        const int k0 = c * KCHUNK;
        if (k0 + 2 * KCHUNK < K) {
            __builtin_prefetch(Wr0 + k0 + 2 * KCHUNK, 0, 1);
            __builtin_prefetch(Wr1 + k0 + 2 * KCHUNK, 0, 1);
        }
        #pragma unroll
        for (int kk = 0; kk < KCHUNK; kk += 32) {
            v16bf a0 = frag_cat(&cur[m * KCHUNK + kk + g * 8],
                                &cur[m * KCHUNK + kk + 16 + g * 8]);
            v16bf a1 = frag_cat(&cur[(16 + m) * KCHUNK + kk + g * 8],
                                &cur[(16 + m) * KCHUNK + kk + 16 + g * 8]);
            v16bf b0 = frag_cat(Wr0 + k0 + kk + g * 8, Wr0 + k0 + kk + 16 + g * 8);
            v16bf b1 = frag_cat(Wr1 + k0 + kk + g * 8, Wr1 + k0 + kk + 16 + g * 8);
            acc[0][0] = wmma_bf16(a0, b0, acc[0][0]);
            acc[1][0] = wmma_bf16(a1, b0, acc[1][0]);
            acc[0][1] = wmma_bf16(a0, b1, acc[0][1]);
            acc[1][1] = wmma_bf16(a1, b1, acc[1][1]);
        }
        if (c + 1 < nchunks) {
            __bf16* nxt = ((c + 1) & 1) ? buf1 : buf0;
            *(v8bf*)&nxt[srow * KCHUNK + scol] = stage;
            if (c + 2 < nchunks)
                stage = *(const v8bf*)(Ag + (size_t)(c + 2) * KCHUNK);
        }
        __syncthreads();
    }
#endif
}

__global__ __launch_bounds__(256) void k_cvt(const float* __restrict__ in,
                                             __bf16* __restrict__ out, int n) {
    int i = blockIdx.x * 256 + threadIdx.x;
    if (i < n) out[i] = (__bf16)in[i];
}

// One block per row; 768 = 3*256.
__global__ __launch_bounds__(256) void k_layernorm(const float* __restrict__ x,
                                                   const float* __restrict__ w,
                                                   const float* __restrict__ b,
                                                   __bf16* __restrict__ out) {
    const int row = blockIdx.x, t = threadIdx.x;
    const size_t base = (size_t)row * DIM;
    float v0 = x[base + t], v1 = x[base + 256 + t], v2 = x[base + 512 + t];
    float s = v0 + v1 + v2, ss = v0 * v0 + v1 * v1 + v2 * v2;
    #pragma unroll
    for (int off = 16; off >= 1; off >>= 1) {
        s  += __shfl_xor(s,  off, 32);
        ss += __shfl_xor(ss, off, 32);
    }
    __shared__ float rs[8], rss[8];
    const int wave = t >> 5, lane = t & 31;
    if (lane == 0) { rs[wave] = s; rss[wave] = ss; }
    __syncthreads();
    s = 0.f; ss = 0.f;
    #pragma unroll
    for (int i = 0; i < 8; ++i) { s += rs[i]; ss += rss[i]; }
    float mu  = s * (1.f / DIM);
    float var = ss * (1.f / DIM) - mu * mu;
    float r   = rsqrtf(var + 1e-5f);
    out[base + t]       = (__bf16)((v0 - mu) * r * w[t]       + b[t]);
    out[base + 256 + t] = (__bf16)((v1 - mu) * r * w[256 + t] + b[256 + t]);
    out[base + 512 + t] = (__bf16)((v2 - mu) * r * w[512 + t] + b[512 + t]);
}

// QKV GEMM: [12608,768] x [2304,768]^T, scatter to [3,B,H,197,64] bf16, q pre-scaled.
__global__ __launch_bounds__(256) void k_gemm_qkv(const __bf16* __restrict__ A,
                                                  const __bf16* __restrict__ W,
                                                  const float* __restrict__ qb,
                                                  const float* __restrict__ vb,
                                                  __bf16* __restrict__ qkv) {
    __shared__ __align__(16) __bf16 lds[3 * 32 * KCHUNK];
    const int wave = threadIdx.x >> 5, lane = threadIdx.x & 31;
    const int m = lane & 15, g = lane >> 4;
    const int m0  = blockIdx.x * 32;
    const int n0w = blockIdx.y * 256 + wave * 32;
    v8f acc[2][2];
    gemm_rowblock32(A, W, DIM, m0, n0w, lds, acc);
    #pragma unroll
    for (int mt = 0; mt < 2; ++mt)
    #pragma unroll
    for (int nt = 0; nt < 2; ++nt)
    #pragma unroll
    for (int i = 0; i < 8; ++i) {
        float val = acc[mt][nt][i];
        int c     = n0w + nt * 16 + m;
        int which = c / DIM;
        int rem   = c - which * DIM;
        if (which == 0)      val = (val + qb[rem]) * QSCALE;
        else if (which == 2) val += vb[rem];
        int grow = m0 + mt * 16 + i + 8 * g;
        int bb = grow / NTOK, nn = grow - bb * NTOK;
        int h = rem >> 6, d = rem & 63;
        size_t dst = ((((size_t)which * BATCH + bb) * HEADS + h) * NTOK + nn)
                         * HEAD_DIM + d;
        qkv[dst] = (__bf16)val;
    }
}

// One wave per (query-tile, head, batch). Scores in LDS f32, softmax with shuffles,
// P bf16 in LDS, P@V via WMMA.
__global__ __launch_bounds__(32) void k_attention(const __bf16* __restrict__ qkv,
                                                  const float* __restrict__ table,
                                                  const int* __restrict__ rel_index,
                                                  __bf16* __restrict__ attn_out) {
    __shared__ __align__(16) float  S[16 * NKPAD];
    __shared__ __align__(16) __bf16 P[16 * NKPAD];
    const int mt = blockIdx.x;       // 0..12
    const int h  = blockIdx.y;       // 0..11
    const int b  = blockIdx.z;       // 0..63
    const int lane = threadIdx.x;
    const int m = lane & 15, g = lane >> 4;
    const size_t per = (size_t)BATCH * HEADS * NTOK * HEAD_DIM;
    const size_t bh  = ((size_t)b * HEADS + h) * NTOK * HEAD_DIM;
    const __bf16* q = qkv + bh;
    const __bf16* k = qkv + per + bh;
    const __bf16* v = qkv + 2 * per + bh;
    const int m0 = mt * 16;

    const int qrow = (m0 + m < NTOK) ? (m0 + m) : (NTOK - 1);
    const __bf16* qr = q + (size_t)qrow * HEAD_DIM;
    v16bf qa0 = frag_cat(qr + g * 8,      qr + 16 + g * 8);
    v16bf qa1 = frag_cat(qr + 32 + g * 8, qr + 48 + g * 8);

    for (int j = 0; j < 13; ++j) {
        int krow = (j * 16 + m < NTOK) ? (j * 16 + m) : (NTOK - 1);
        const __bf16* kr = k + (size_t)krow * HEAD_DIM;
        v16bf kb0 = frag_cat(kr + g * 8,      kr + 16 + g * 8);
        v16bf kb1 = frag_cat(kr + 32 + g * 8, kr + 48 + g * 8);
        v8f acc = {};
        acc = wmma_bf16(qa0, kb0, acc);
        acc = wmma_bf16(qa1, kb1, acc);
        #pragma unroll
        for (int i = 0; i < 8; ++i) {
            int qm = m0 + i + 8 * g;
            int kn = j * 16 + m;
            float sv;
            if (qm < NTOK && kn < NTOK) {
                int ridx = rel_index[qm * NTOK + kn];
                sv = acc[i] + table[ridx * HEADS + h];
            } else {
                sv = -1e30f;
            }
            S[(i + 8 * g) * NKPAD + j * 16 + m] = sv;
        }
    }
    for (int idx = lane; idx < 16 * 16; idx += 32) {        // pad cols 208..223
        S[(idx >> 4) * NKPAD + 208 + (idx & 15)] = -1e30f;
    }
    __syncthreads();

    for (int r = 0; r < 16; ++r) {                          // row softmax
        float vals[7];
        float mx = -1e30f;
        #pragma unroll
        for (int c = 0; c < 7; ++c) {
            vals[c] = S[r * NKPAD + c * 32 + lane];
            mx = fmaxf(mx, vals[c]);
        }
        #pragma unroll
        for (int off = 16; off >= 1; off >>= 1) mx = fmaxf(mx, __shfl_xor(mx, off, 32));
        float sum = 0.f;
        #pragma unroll
        for (int c = 0; c < 7; ++c) { vals[c] = __expf(vals[c] - mx); sum += vals[c]; }
        #pragma unroll
        for (int off = 16; off >= 1; off >>= 1) sum += __shfl_xor(sum, off, 32);
        float inv = 1.0f / sum;
        #pragma unroll
        for (int c = 0; c < 7; ++c)
            P[r * NKPAD + c * 32 + lane] = (__bf16)(vals[c] * inv);
    }
    __syncthreads();

    #pragma unroll
    for (int nt = 0; nt < 4; ++nt) {                        // O = P @ V, 64 cols
        v8f acc = {};
        for (int ks = 0; ks < 7; ++ks) {
            v16bf pa = frag_cat(&P[m * NKPAD + ks * 32 + g * 8],
                                &P[m * NKPAD + ks * 32 + 16 + g * 8]);
            v16bf vbf;
            #pragma unroll
            for (int e = 0; e < 8; ++e) {
                int kr0 = ks * 32 + g * 8 + e;
                int kr1 = ks * 32 + 16 + g * 8 + e;
                kr0 = (kr0 < NTOK) ? kr0 : (NTOK - 1);      // padded rows * P==0
                kr1 = (kr1 < NTOK) ? kr1 : (NTOK - 1);
                vbf[e]     = v[(size_t)kr0 * HEAD_DIM + nt * 16 + m];
                vbf[e + 8] = v[(size_t)kr1 * HEAD_DIM + nt * 16 + m];
            }
            acc = wmma_bf16(pa, vbf, acc);
        }
        #pragma unroll
        for (int i = 0; i < 8; ++i) {
            int qm = m0 + i + 8 * g;
            if (qm < NTOK)
                attn_out[((size_t)b * NTOK + qm) * DIM + h * HEAD_DIM + nt * 16 + m] =
                    (__bf16)acc[i];
        }
    }
}

__global__ __launch_bounds__(256) void k_gemm_proj(const __bf16* __restrict__ A,
                                                   const __bf16* __restrict__ W,
                                                   const float* __restrict__ bias,
                                                   const float* __restrict__ x,
                                                   const float* __restrict__ gamma,
                                                   float* __restrict__ out1) {
    __shared__ __align__(16) __bf16 lds[3 * 32 * KCHUNK];
    const int wave = threadIdx.x >> 5, lane = threadIdx.x & 31;
    const int m = lane & 15, g = lane >> 4;
    const int m0  = blockIdx.x * 32;
    const int n0w = blockIdx.y * 256 + wave * 32;
    v8f acc[2][2];
    gemm_rowblock32(A, W, DIM, m0, n0w, lds, acc);
    #pragma unroll
    for (int mt = 0; mt < 2; ++mt)
    #pragma unroll
    for (int nt = 0; nt < 2; ++nt)
    #pragma unroll
    for (int i = 0; i < 8; ++i) {
        int c = n0w + nt * 16 + m;
        int grow = m0 + mt * 16 + i + 8 * g;
        size_t o = (size_t)grow * DIM + c;
        out1[o] = x[o] + gamma[c] * (acc[mt][nt][i] + bias[c]);
    }
}

__global__ __launch_bounds__(256) void k_gemm_fc1(const __bf16* __restrict__ A,
                                                  const __bf16* __restrict__ W,
                                                  const float* __restrict__ bias,
                                                  __bf16* __restrict__ hmid) {
    __shared__ __align__(16) __bf16 lds[3 * 32 * KCHUNK];
    const int wave = threadIdx.x >> 5, lane = threadIdx.x & 31;
    const int m = lane & 15, g = lane >> 4;
    const int m0  = blockIdx.x * 32;
    const int n0w = blockIdx.y * 256 + wave * 32;
    v8f acc[2][2];
    gemm_rowblock32(A, W, DIM, m0, n0w, lds, acc);
    #pragma unroll
    for (int mt = 0; mt < 2; ++mt)
    #pragma unroll
    for (int nt = 0; nt < 2; ++nt)
    #pragma unroll
    for (int i = 0; i < 8; ++i) {
        int c = n0w + nt * 16 + m;
        float val = acc[mt][nt][i] + bias[c];
        val = 0.5f * val * (1.0f + erff(val * 0.70710678f));  // exact GELU
        int grow = m0 + mt * 16 + i + 8 * g;
        hmid[(size_t)grow * HIDDEN + c] = (__bf16)val;
    }
}

__global__ __launch_bounds__(256) void k_gemm_fc2(const __bf16* __restrict__ A,
                                                  const __bf16* __restrict__ W,
                                                  const float* __restrict__ bias,
                                                  const float* __restrict__ out1,
                                                  const float* __restrict__ gamma,
                                                  float* __restrict__ out) {
    __shared__ __align__(16) __bf16 lds[3 * 32 * KCHUNK];
    const int wave = threadIdx.x >> 5, lane = threadIdx.x & 31;
    const int m = lane & 15, g = lane >> 4;
    const int m0  = blockIdx.x * 32;
    const int n0w = blockIdx.y * 256 + wave * 32;
    v8f acc[2][2];
    gemm_rowblock32(A, W, HIDDEN, m0, n0w, lds, acc);
    #pragma unroll
    for (int mt = 0; mt < 2; ++mt)
    #pragma unroll
    for (int nt = 0; nt < 2; ++nt)
    #pragma unroll
    for (int i = 0; i < 8; ++i) {
        int c = n0w + nt * 16 + m;
        int grow = m0 + mt * 16 + i + 8 * g;
        size_t o = (size_t)grow * DIM + c;
        out[o] = out1[o] + gamma[c] * (acc[mt][nt][i] + bias[c]);
    }
}

extern "C" void kernel_launch(void* const* d_in, const int* in_sizes, int n_in,
                              void* d_out, int out_size, void* d_ws, size_t ws_size,
                              hipStream_t stream) {
    (void)in_sizes; (void)n_in; (void)out_size; (void)ws_size;
    const float* x      = (const float*)d_in[0];
    const float* qkv_w  = (const float*)d_in[1];
    const float* q_bias = (const float*)d_in[2];
    const float* v_bias = (const float*)d_in[3];
    const float* table  = (const float*)d_in[4];
    const float* proj_w = (const float*)d_in[5];
    const float* proj_b = (const float*)d_in[6];
    const float* n1_w   = (const float*)d_in[7];
    const float* n1_b   = (const float*)d_in[8];
    const float* n2_w   = (const float*)d_in[9];
    const float* n2_b   = (const float*)d_in[10];
    const float* fc1_w  = (const float*)d_in[11];
    const float* fc1_b  = (const float*)d_in[12];
    const float* fc2_w  = (const float*)d_in[13];
    const float* fc2_b  = (const float*)d_in[14];
    const float* gamma1 = (const float*)d_in[15];
    const float* gamma2 = (const float*)d_in[16];
    const int* rel_idx  = (const int*)d_in[17];
    float* out = (float*)d_out;

    uintptr_t p = (uintptr_t)d_ws;
    auto alloc = [&](size_t bytes) -> void* {
        uintptr_t r = p;
        p += (bytes + 255) & ~(size_t)255;
        return (void*)r;
    };
    __bf16* w_qkv  = (__bf16*)alloc((size_t)3 * DIM * DIM * 2);
    __bf16* w_proj = (__bf16*)alloc((size_t)DIM * DIM * 2);
    __bf16* w_fc1  = (__bf16*)alloc((size_t)HIDDEN * DIM * 2);
    __bf16* w_fc2  = (__bf16*)alloc((size_t)DIM * HIDDEN * 2);
    __bf16* ln1    = (__bf16*)alloc((size_t)NROWS * DIM * 2);
    __bf16* qkvb   = (__bf16*)alloc((size_t)3 * BATCH * HEADS * NTOK * HEAD_DIM * 2);
    __bf16* attnb  = (__bf16*)alloc((size_t)NROWS * DIM * 2);
    float*  out1   = (float*)alloc((size_t)NROWS * DIM * 4);
    __bf16* ln2    = (__bf16*)alloc((size_t)NROWS * DIM * 2);
    __bf16* hmid   = (__bf16*)alloc((size_t)NROWS * HIDDEN * 2);

    int n;
    n = 3 * DIM * DIM;  k_cvt<<<(n + 255) / 256, 256, 0, stream>>>(qkv_w, w_qkv, n);
    n = DIM * DIM;      k_cvt<<<(n + 255) / 256, 256, 0, stream>>>(proj_w, w_proj, n);
    n = HIDDEN * DIM;   k_cvt<<<(n + 255) / 256, 256, 0, stream>>>(fc1_w, w_fc1, n);
    n = DIM * HIDDEN;   k_cvt<<<(n + 255) / 256, 256, 0, stream>>>(fc2_w, w_fc2, n);

    k_layernorm<<<NROWS, 256, 0, stream>>>(x, n1_w, n1_b, ln1);
    k_gemm_qkv<<<dim3(MTILES / 2, 9), 256, 0, stream>>>(ln1, w_qkv, q_bias, v_bias, qkvb);
    k_attention<<<dim3(13, HEADS, BATCH), 32, 0, stream>>>(qkvb, table, rel_idx, attnb);
    k_gemm_proj<<<dim3(MTILES / 2, 3), 256, 0, stream>>>(attnb, w_proj, proj_b, x, gamma1, out1);
    k_layernorm<<<NROWS, 256, 0, stream>>>(out1, n2_w, n2_b, ln2);
    k_gemm_fc1<<<dim3(MTILES / 2, 12), 256, 0, stream>>>(ln2, w_fc1, fc1_b, hmid);
    k_gemm_fc2<<<dim3(MTILES / 2, 3), 256, 0, stream>>>(hmid, w_fc2, fc2_b, out1, gamma2, out);
}
